// MultiHeadAttention_90099823935648
// MI455X (gfx1250) — matrix operand order repaired and checked by reading, not compiled, not run on hardware
//
#include <hip/hip_runtime.h>
#include <hip/hip_bf16.h>
#include <cstdint>
#include <cstddef>

// ---------------------------------------------------------------------------
// MI455X (gfx1250) MHA forward.
// Precision: V_WMMA_F32_16X16X32_BF16 (bf16 A/B, f32 accum); softmax in f32.
// Roofline: ~553MB mandatory output (att weights dominate) => ~28us @23.3TB/s;
// 103 GFLOP needs >3.7 PFLOPS to stay memory-bound => bf16 matrix path.
// The 536MB att write is emitted as fully-coalesced float4 stores.
// ---------------------------------------------------------------------------

typedef __attribute__((ext_vector_type(16))) __bf16 v16bf;
typedef __attribute__((ext_vector_type(8)))  __bf16 v8bf;
typedef __attribute__((ext_vector_type(8)))  float  v8f;

#define WMMA_BF16(A_, B_, C_) \
  __builtin_amdgcn_wmma_f32_16x16x32_bf16(false, (A_), false, (B_), (short)0, (C_), false, false)

// constants
#define NHEADS 16
#define DMODEL 1024
#define DEPTH  64
#define BATCH  2
#define SEQ    2048
#define ROWS   (BATCH * SEQ)   // 4096

// ---------------------------------------------------------------------------
// A-fragment builders (16x32, ISA 7.12.2: lane=row (lane&15), half=lane>>4,
// callers give elems 0..15 -> K = 16*half + e (contiguous, same as B; not the ISA order))
// ---------------------------------------------------------------------------
static __device__ __forceinline__ v16bf a_from_f32(const float* p0, const float* p1) {
  float4 x0 = ((const float4*)p0)[0];
  float4 x1 = ((const float4*)p0)[1];
  float4 y0 = ((const float4*)p1)[0];
  float4 y1 = ((const float4*)p1)[1];
  v16bf a;
  a[0]  = (__bf16)x0.x; a[1]  = (__bf16)x0.y; a[2]  = (__bf16)x0.z; a[3]  = (__bf16)x0.w;
  a[4]  = (__bf16)x1.x; a[5]  = (__bf16)x1.y; a[6]  = (__bf16)x1.z; a[7]  = (__bf16)x1.w;
  a[8]  = (__bf16)y0.x; a[9]  = (__bf16)y0.y; a[10] = (__bf16)y0.z; a[11] = (__bf16)y0.w;
  a[12] = (__bf16)y1.x; a[13] = (__bf16)y1.y; a[14] = (__bf16)y1.z; a[15] = (__bf16)y1.w;
  return a;
}

static __device__ __forceinline__ v16bf a_from_bf16(const __bf16* p0, const __bf16* p1) {
  v8bf x = *(const v8bf*)p0;
  v8bf y = *(const v8bf*)p1;
  v16bf a;
#pragma unroll
  for (int i = 0; i < 8; ++i) { a[i] = x[i]; a[8 + i] = y[i]; }
  return a;
}

// ---------------------------------------------------------------------------
// Weight transpose + f32->bf16:  W[k][n] (row-major f32) -> Wt[n][k] bf16
// LDS-tiled (64x65 bf16) so both read and write sides are coalesced.
// Grid: (16, 16) tiles of 64x64, 256 threads.
// ---------------------------------------------------------------------------
__global__ __launch_bounds__(256) void wt_bf16_kernel(const float* __restrict__ W,
                                                      __bf16* __restrict__ Wt) {
  __shared__ __bf16 tile[64][65];
  const int tx = threadIdx.x & 63;        // contiguous dim
  const int tg = threadIdx.x >> 6;        // 0..3
  const int n0 = blockIdx.x * 64;
  const int k0 = blockIdx.y * 64;

#pragma unroll 4
  for (int i = 0; i < 16; ++i) {
    const int r = tg * 16 + i;            // row within tile
    // read W[k0+r][n0+tx] : coalesced in tx
    tile[tx][r] = (__bf16)W[(size_t)(k0 + r) * DMODEL + n0 + tx];
  }
  __syncthreads();
#pragma unroll 4
  for (int i = 0; i < 16; ++i) {
    const int r = tg * 16 + i;
    // write Wt[n0+r][k0+tx] = W[k0+tx][n0+r] = tile[r][tx] : coalesced in tx
    Wt[(size_t)(n0 + r) * DMODEL + k0 + tx] = tile[r][tx];
  }
}

// ---------------------------------------------------------------------------
// Tiled WMMA GEMM: Y[R=4096][1024] = A[4096][1024] @ Wt^T + bias
//   Bt stored N-major: Bt[n][k] -> B-fragment is one 32-byte contiguous load.
// MODE 0: store bf16 [B,H,S,64]     (q/k head split)
// MODE 1: store bf16 [B,H,64,S]     (v head split, transposed for ctx GEMM)
// MODE 2: store f32 row-major       (final output)
// ---------------------------------------------------------------------------
template <int MODE, bool ABF16>
__global__ __launch_bounds__(128) void gemm_wmma_kernel(const void* __restrict__ Aptr,
                                                        const __bf16* __restrict__ Bt,
                                                        const float* __restrict__ bias,
                                                        void* __restrict__ out) {
  const int lane = threadIdx.x & 31;
  const int wave = threadIdx.x >> 5;
  const int row  = lane & 15;
  const int half = lane >> 4;
  const int n0 = blockIdx.x * 64;
  const int m0 = blockIdx.y * 64 + wave * 16;

  v8f acc[4] = {};

  for (int kb = 0; kb < DMODEL; kb += 32) {
    v16bf a;
    if (ABF16) {
      const __bf16* ar = (const __bf16*)Aptr + (size_t)(m0 + row) * DMODEL + kb;
      a = a_from_bf16(ar + 16 * half, ar + 8 + 16 * half);
    } else {
      const float* ar = (const float*)Aptr + (size_t)(m0 + row) * DMODEL + kb;
      a = a_from_f32(ar + 16 * half, ar + 8 + 16 * half);
    }
#pragma unroll
    for (int g = 0; g < 4; ++g) {
      const __bf16* bp = Bt + (size_t)(n0 + g * 16 + row) * DMODEL + kb + 16 * half;
      v16bf bf = *(const v16bf*)bp;
      acc[g] = WMMA_BF16(a, bf, acc[g]);
    }
  }

#pragma unroll
  for (int g = 0; g < 4; ++g) {
    const int col = n0 + g * 16 + row;
    const float bv = bias[col];
#pragma unroll
    for (int i = 0; i < 8; ++i) {
      const int R = m0 + i + 8 * half;   // C layout: M = i + 8*half, N = lane&15
      const float val = acc[g][i] + bv;
      if (MODE == 2) {
        ((float*)out)[(size_t)R * DMODEL + col] = val;
      } else {
        const int b = R >> 11, s = R & (SEQ - 1);
        const int h = col >> 6, d = col & (DEPTH - 1);
        const __bf16 vb = (__bf16)val;
        if (MODE == 0)
          ((__bf16*)out)[(((size_t)(b * NHEADS + h)) * SEQ + s) * DEPTH + d] = vb;
        else
          ((__bf16*)out)[(((size_t)(b * NHEADS + h)) * DEPTH + d) * SEQ + s] = vb;
      }
    }
  }
}

// ---------------------------------------------------------------------------
// Attention: one workgroup (4 waves / 128 thr) per (b, h, 16-query tile).
//  Phase 1: scores(16 x 2048) = qh @ kh^T * 1/8  -> 128KB f32 LDS tile (WMMA)
//  Phase 2: f32 softmax over keys, THEN + mask*(-1e9) (faithful to reference);
//           att weights written as coalesced float4 (global_store_b128),
//           masked weights kept in LDS
//  Phase 3: ctx(16 x 64) = att @ vh via WMMA (LDS f32 -> bf16 A on the fly,
//           vhT gives contiguous-K B-fragments), store ctx bf16 [B,S,1024]
// ---------------------------------------------------------------------------
__global__ __launch_bounds__(128) void attention_kernel(const __bf16* __restrict__ qh,
                                                        const __bf16* __restrict__ kh,
                                                        const __bf16* __restrict__ vhT,
                                                        const float* __restrict__ mask,
                                                        float* __restrict__ att_out,
                                                        __bf16* __restrict__ ctx) {
  extern __shared__ float4 lds4[];
  float* scores = (float*)lds4;              // [16][2048]
  float* red    = scores + 16 * SEQ;         // [16][8] partial reductions
  float* rinv   = red + 128;                 // [16] per-row 1/sum

  const int qt = blockIdx.x;                 // 0..127
  const int h  = blockIdx.y;                 // 0..15
  const int b  = blockIdx.z;                 // 0..1
  const int bh = b * NHEADS + h;
  const int q0 = qt * 16;

  const int tid  = threadIdx.x;
  const int lane = tid & 31;
  const int wave = tid >> 5;
  const int row  = lane & 15;
  const int half = lane >> 4;

  // ---- Phase 1: scores ----
  const __bf16* qrow = qh + ((size_t)bh * SEQ + q0) * DEPTH + (size_t)row * DEPTH;
  v16bf a0 = a_from_bf16(qrow +      16 * half, qrow +  8 + 16 * half);   // depth 0..31
  v16bf a1 = a_from_bf16(qrow + 32 + 16 * half, qrow + 40 + 16 * half);   // depth 32..63

  for (int kt = wave; kt < SEQ / 16; kt += 4) {
    const int k0 = kt * 16;
    const __bf16* krow = kh + ((size_t)bh * SEQ + k0) * DEPTH + (size_t)row * DEPTH;
    v16bf b0 = *(const v16bf*)(krow +      16 * half);                  // depth 0..31
    v16bf b1 = *(const v16bf*)(krow + 32 + 16 * half);                  // depth 32..63
    v8f c = {};
    c = WMMA_BF16(a0, b0, c);
    c = WMMA_BF16(a1, b1, c);
#pragma unroll
    for (int i = 0; i < 8; ++i)
      scores[(size_t)(i + 8 * half) * SEQ + k0 + row] = c[i] * 0.125f;  // 1/sqrt(64)
  }
  __syncthreads();

  // ---- Phase 2: softmax (f32), vectorized float4 passes ----
  const int r   = tid >> 3;   // row 0..15
  const int seg = tid & 7;    // 256-element (64 x float4) segment
  float4* s4 = (float4*)(scores + (size_t)r * SEQ + seg * 256);

  float m = -3.4e38f;
#pragma unroll 4
  for (int j = 0; j < 64; ++j) {
    float4 x = s4[j];
    m = fmaxf(m, fmaxf(fmaxf(x.x, x.y), fmaxf(x.z, x.w)));
  }
  red[r * 8 + seg] = m;
  __syncthreads();
  float rmax = red[r * 8];
#pragma unroll
  for (int t = 1; t < 8; ++t) rmax = fmaxf(rmax, red[r * 8 + t]);
  __syncthreads();

  float s = 0.0f;
#pragma unroll 4
  for (int j = 0; j < 64; ++j) {
    float4 x = s4[j];
    x.x = __expf(x.x - rmax); x.y = __expf(x.y - rmax);
    x.z = __expf(x.z - rmax); x.w = __expf(x.w - rmax);
    s4[j] = x;
    s += (x.x + x.y) + (x.z + x.w);
  }
  red[r * 8 + seg] = s;
  __syncthreads();
  float rsum = 0.0f;
#pragma unroll
  for (int t = 0; t < 8; ++t) rsum += red[r * 8 + t];
  if (seg == 0) rinv[r] = 1.0f / rsum;
  __syncthreads();

  // normalize + post-softmax mask; att written as coalesced float4 stores:
  // consecutive lanes -> consecutive 16B chunks within a 2048-wide row.
  const float* mask_b = mask + (size_t)b * SEQ;
  float* att_base = att_out + ((size_t)bh * SEQ + q0) * SEQ;
  for (int idx = tid * 4; idx < 16 * SEQ; idx += 128 * 4) {
    const int rr = idx >> 11;            // row 0..15
    const int cc = idx & (SEQ - 1);      // col
    float4 e  = *(float4*)(scores + (size_t)rr * SEQ + cc);
    float4 mk = *(const float4*)(mask_b + cc);
    const float iv = rinv[rr];
    float4 v;
    v.x = e.x * iv + mk.x * (-1e9f);
    v.y = e.y * iv + mk.y * (-1e9f);
    v.z = e.z * iv + mk.z * (-1e9f);
    v.w = e.w * iv + mk.w * (-1e9f);
    *(float4*)(att_base + (size_t)rr * SEQ + cc) = v;   // global_store_b128
    *(float4*)(scores + (size_t)rr * SEQ + cc) = v;     // keep for ctx GEMM
  }
  __syncthreads();

  // ---- Phase 3: ctx = att @ vh ; wave handles depth cols [16*wave, +16) ----
  const int d0 = wave * 16;
  v8f c = {};
  for (int kb = 0; kb < SEQ; kb += 32) {
    const float* p0 = scores + (size_t)row * SEQ + kb + 16 * half;
    v16bf a = a_from_f32(p0, p0 + 8);
    const __bf16* vp = vhT + ((size_t)bh * DEPTH + d0 + row) * SEQ + kb + 16 * half;
    v16bf bf = *(const v16bf*)vp;
    c = WMMA_BF16(a, bf, c);
  }
#pragma unroll
  for (int i = 0; i < 8; ++i) {
    const int q = q0 + i + 8 * half;
    const int col = h * DEPTH + d0 + row;
    ctx[((size_t)b * SEQ + q) * DMODEL + col] = (__bf16)c[i];
  }
}

// ---------------------------------------------------------------------------
// Host launch
// ---------------------------------------------------------------------------
extern "C" void kernel_launch(void* const* d_in, const int* in_sizes, int n_in,
                              void* d_out, int out_size, void* d_ws, size_t ws_size,
                              hipStream_t stream) {
  (void)in_sizes; (void)n_in; (void)out_size; (void)ws_size;

  const float* q_in  = (const float*)d_in[0];
  const float* k_in  = (const float*)d_in[1];
  const float* v_in  = (const float*)d_in[2];
  const float* mask  = (const float*)d_in[3];
  const float* Wq    = (const float*)d_in[4];
  const float* bq    = (const float*)d_in[5];
  const float* Wk    = (const float*)d_in[6];
  const float* bk    = (const float*)d_in[7];
  const float* Wv    = (const float*)d_in[8];
  const float* bv    = (const float*)d_in[9];
  const float* Wo    = (const float*)d_in[10];
  const float* bo    = (const float*)d_in[11];

  float* out0 = (float*)d_out;                                   // [B,S,1024] f32
  float* att  = (float*)d_out + (size_t)ROWS * DMODEL;           // [B,H,S,S]  f32

  // workspace layout (40 MB total)
  char* ws = (char*)d_ws;
  __bf16* WqT = (__bf16*)(ws + 0);                // 2MB each, Wt[n][k]
  __bf16* WkT = (__bf16*)(ws + (1u << 21));
  __bf16* WvT = (__bf16*)(ws + (2u << 21));
  __bf16* WoT = (__bf16*)(ws + (3u << 21));
  __bf16* qh  = (__bf16*)(ws + (4u << 21));       // 8MB  [B,H,S,64]
  __bf16* kh  = (__bf16*)(ws + (8u << 21));       // 8MB  [B,H,S,64]
  __bf16* vhT = (__bf16*)(ws + (12u << 21));      // 8MB  [B,H,64,S]
  __bf16* ctx = (__bf16*)(ws + (16u << 21));      // 8MB  [B,S,1024]

  // 1) weight transpose+convert (LDS-tiled, coalesced both sides)
  dim3 tgrid(DMODEL / 64, DMODEL / 64);
  wt_bf16_kernel<<<tgrid, 256, 0, stream>>>(Wq, WqT);
  wt_bf16_kernel<<<tgrid, 256, 0, stream>>>(Wk, WkT);
  wt_bf16_kernel<<<tgrid, 256, 0, stream>>>(Wv, WvT);
  wt_bf16_kernel<<<tgrid, 256, 0, stream>>>(Wo, WoT);

  // 2) projections (WMMA bf16): grid = (1024/64, 4096/64)
  dim3 ggrid(DMODEL / 64, ROWS / 64);
  gemm_wmma_kernel<0, false><<<ggrid, 128, 0, stream>>>((const void*)q_in, WqT, bq, (void*)qh);
  gemm_wmma_kernel<0, false><<<ggrid, 128, 0, stream>>>((const void*)k_in, WkT, bk, (void*)kh);
  gemm_wmma_kernel<1, false><<<ggrid, 128, 0, stream>>>((const void*)v_in, WvT, bv, (void*)vhT);

  // 3) attention: grid (S/16, H, B), ~128.6KB dynamic LDS per workgroup
  const size_t smem = (size_t)(16 * SEQ + 128 + 16) * sizeof(float);
  hipFuncSetAttribute((const void*)attention_kernel,
                      hipFuncAttributeMaxDynamicSharedMemorySize, (int)smem);
  dim3 agrid(SEQ / 16, NHEADS, BATCH);
  attention_kernel<<<agrid, 128, smem, stream>>>(qh, kh, vhT, mask, att, ctx);

  // 4) output projection (WMMA bf16 A from ctx), f32 store
  gemm_wmma_kernel<2, true><<<ggrid, 128, 0, stream>>>((const void*)ctx, WoT, bo, (void*)out0);
}
